// ScaledDotProductAttention_91311004713070
// MI455X (gfx1250) — matrix operand order repaired and checked by reading, not compiled, not run on hardware
//
#include <hip/hip_runtime.h>

// ---------------------------------------------------------------------------
// Causal self-attention with full-width (d_model) attention dim, for gfx1250.
// All matmuls via v_wmma_f32_16x16x32_f16 (f16 operands, f32 accumulate).
// Compute-bound: ~0.9 TFLOP vs <1 GB of HBM traffic on a 23.3 TB/s part.
// ---------------------------------------------------------------------------

#define DM   1024
#define SEQ  2048
#define NB   32
#define ROWS (NB * SEQ)          // 65536 fused (batch,seq) rows

typedef __attribute__((ext_vector_type(16))) _Float16 v16h;
typedef __attribute__((ext_vector_type(8)))  _Float16 v8h;
typedef __attribute__((ext_vector_type(8)))  float    v8f;

union ABFrag { v16h v; v8h h[2]; _Float16 e[16]; };

// ---- WMMA fragment loaders (ISA 7.12.2 wave32 layouts) ---------------------
// A (16x32, f16): lane m = lane%16; halves 0..7 -> K = kb..kb+7,
// halves 8..15 -> K = kb+16..kb+23, kb = (lane>=16) ? 8 : 0.
__device__ __forceinline__ v16h load_a_f16(const _Float16* base, int ld,
                                           int row0, int k0, int lane) {
  ABFrag u;
  const _Float16* p = base + (size_t)(row0 + (lane & 15)) * ld
                           + (k0 + ((lane & 16) ? 8 : 0));
  u.h[0] = *(const v8h*)(p);
  u.h[1] = *(const v8h*)(p + 16);
  return u.v;
}

// Same A layout but source is fp32 (converted to f16 in registers).
__device__ __forceinline__ v16h load_a_f32(const float* base, int ld,
                                           int row0, int k0, int lane) {
  ABFrag u;
  const float* p = base + (size_t)(row0 + (lane & 15)) * ld
                        + (k0 + ((lane & 16) ? 8 : 0));
#pragma unroll
  for (int i = 0; i < 8; ++i) {
    u.e[i]     = (_Float16)p[i];
    u.e[i + 8] = (_Float16)p[i + 16];
  }
  return u.v;
}

// B (32x16, f16): lane n = lane%16; halves 0..7 -> K = kb..kb+7,
// halves 8..15 -> K = kb+16..kb+23, kb = (lane>=16) ? 8 : 0. Source must be "k-contiguous per n" (ld = k stride).
__device__ __forceinline__ v16h load_b_f16(const _Float16* baseT, int ld,
                                           int n0, int k0, int lane) {
  ABFrag u;
  const _Float16* p = baseT + (size_t)(n0 + (lane & 15)) * ld
                            + (k0 + ((lane & 16) ? 8 : 0));
  u.h[0] = *(const v8h*)(p);
  u.h[1] = *(const v8h*)(p + 16);
  return u.v;
}

__device__ __forceinline__ v8f wmma16(v16h a, v16h b, v8f c) {
  return __builtin_amdgcn_wmma_f32_16x16x32_f16(false, a, false, b,
                                                (short)0, c, false, false);
}

// ---------------------------------------------------------------------------
// Kernel 0: W[d][n] fp32  ->  Wt[n][d] f16   (3 matrices)
// ---------------------------------------------------------------------------
__global__ __launch_bounds__(256) void wconvert_kernel(const float* __restrict__ Wq,
                                                       const float* __restrict__ Wk,
                                                       const float* __restrict__ Wv,
                                                       _Float16* __restrict__ Wt) {
  size_t idx = (size_t)blockIdx.x * 256 + threadIdx.x;      // 3 * 2^20 total
  int which = (int)(idx >> 20);
  int rem   = (int)(idx & ((1u << 20) - 1));
  int d = rem >> 10;          // n fastest -> coalesced reads of W[d][*]
  int n = rem & (DM - 1);
  const float* W = (which == 0) ? Wq : (which == 1) ? Wk : Wv;
  Wt[((size_t)which << 20) + (size_t)n * DM + d] = (_Float16)W[(size_t)d * DM + n];
}

// ---------------------------------------------------------------------------
// Kernel 1: QKV projection.  grid = (ROWS/128, DM/128, 3)
// 4 waves/block, 64x64 tile per wave, K loop over DM in steps of 32.
// which==0 -> Qh (scaled by 1/sqrt(DM)), which==1 -> Kh, which==2 -> Vt[b][n][t]
// ---------------------------------------------------------------------------
__global__ __launch_bounds__(128) void qkv_gemm_kernel(const float* __restrict__ x,
                                                       const _Float16* __restrict__ Wt,
                                                       const float* __restrict__ bq,
                                                       const float* __restrict__ bk,
                                                       const float* __restrict__ bv,
                                                       _Float16* __restrict__ Qh,
                                                       _Float16* __restrict__ Kh,
                                                       _Float16* __restrict__ Vt) {
  const int lane = threadIdx.x & 31;
  const int wave = threadIdx.x >> 5;
  const int which = blockIdx.z;
  const int m0 = blockIdx.x * 128 + (wave >> 1) * 64;
  const int n0 = blockIdx.y * 128 + (wave & 1) * 64;
  const _Float16* Wsel = Wt + ((size_t)which << 20);
  const float* bias = (which == 0) ? bq : (which == 1) ? bk : bv;

  v8f acc[4][4] = {};
  for (int k0 = 0; k0 < DM; k0 += 32) {
    v16h af[4], bf[4];
#pragma unroll
    for (int mi = 0; mi < 4; ++mi)
      af[mi] = load_a_f32(x, DM, m0 + mi * 16, k0, lane);
#pragma unroll
    for (int ni = 0; ni < 4; ++ni)
      bf[ni] = load_b_f16(Wsel, DM, n0 + ni * 16, k0, lane);
    if (k0 + 64 < DM) {   // CDNA5 global_prefetch_b8 for next-next k-step
      __builtin_prefetch(x    + (size_t)(m0 + (lane & 15)) * DM + k0 + 64, 0, 1);
      __builtin_prefetch(Wsel + (size_t)(n0 + (lane & 15)) * DM + k0 + 64, 0, 1);
    }
#pragma unroll
    for (int mi = 0; mi < 4; ++mi)
#pragma unroll
      for (int ni = 0; ni < 4; ++ni)
        acc[mi][ni] = wmma16(af[mi], bf[ni], acc[mi][ni]);
  }

  const float qscale = 0.03125f;  // 1/sqrt(1024)
#pragma unroll
  for (int mi = 0; mi < 4; ++mi) {
#pragma unroll
    for (int ni = 0; ni < 4; ++ni) {
      const int col = n0 + ni * 16 + (lane & 15);
      const int rbase = m0 + mi * 16 + ((lane & 16) ? 8 : 0);
      const float bval = bias[col];
#pragma unroll
      for (int r = 0; r < 8; ++r) {
        const int row = rbase + r;                 // fused (b,s) row
        float v = acc[mi][ni][r] + bval;
        if (which == 0) v *= qscale;
        if (which < 2) {
          _Float16* dst = (which == 0) ? Qh : Kh;
          dst[(size_t)row * DM + col] = (_Float16)v;
        } else {
          const int b = row >> 11;                 // SEQ == 2048
          const int t = row & (SEQ - 1);
          Vt[((size_t)b * DM + col) * SEQ + t] = (_Float16)v;
        }
      }
    }
  }
}

// ---------------------------------------------------------------------------
// Kernel 2: scores P = (Q * 1/sqrt(D)) @ K^T with causal mask, f16 out.
// grid = (SEQ/128 s-tiles, SEQ/128 t-tiles, NB). Upper blocks skipped.
// ---------------------------------------------------------------------------
__global__ __launch_bounds__(128) void scores_gemm_kernel(const _Float16* __restrict__ Qh,
                                                          const _Float16* __restrict__ Kh,
                                                          _Float16* __restrict__ P) {
  const int s_blk = blockIdx.x * 128;
  const int t_blk = blockIdx.y * 128;
  if (t_blk > s_blk) return;                       // never read by pv_gemm
  const int b = blockIdx.z;
  const int lane = threadIdx.x & 31;
  const int wave = threadIdx.x >> 5;
  const int m0 = s_blk + (wave >> 1) * 64;
  const int n0 = t_blk + (wave & 1) * 64;
  const _Float16* Qb = Qh + (size_t)b * SEQ * DM;
  const _Float16* Kb = Kh + (size_t)b * SEQ * DM;

  v8f acc[4][4] = {};
  for (int k0 = 0; k0 < DM; k0 += 32) {
    v16h af[4], bf[4];
#pragma unroll
    for (int mi = 0; mi < 4; ++mi)
      af[mi] = load_a_f16(Qb, DM, m0 + mi * 16, k0, lane);
#pragma unroll
    for (int ni = 0; ni < 4; ++ni)
      bf[ni] = load_b_f16(Kb, DM, n0 + ni * 16, k0, lane);
    if (k0 + 64 < DM) {
      __builtin_prefetch(Qb + (size_t)(m0 + (lane & 15)) * DM + k0 + 64, 0, 1);
      __builtin_prefetch(Kb + (size_t)(n0 + (lane & 15)) * DM + k0 + 64, 0, 1);
    }
#pragma unroll
    for (int mi = 0; mi < 4; ++mi)
#pragma unroll
      for (int ni = 0; ni < 4; ++ni)
        acc[mi][ni] = wmma16(af[mi], bf[ni], acc[mi][ni]);
  }

  _Float16* Pb = P + (size_t)b * SEQ * SEQ;
  const _Float16 ninf = (_Float16)(-__builtin_inff());
#pragma unroll
  for (int mi = 0; mi < 4; ++mi) {
#pragma unroll
    for (int ni = 0; ni < 4; ++ni) {
      const int col = n0 + ni * 16 + (lane & 15);              // key index t
      const int rbase = m0 + mi * 16 + ((lane & 16) ? 8 : 0);  // query index s
#pragma unroll
      for (int r = 0; r < 8; ++r) {
        const int row = rbase + r;
        _Float16 hv = (col <= row) ? (_Float16)acc[mi][ni][r] : ninf;
        Pb[(size_t)row * SEQ + col] = hv;
      }
    }
  }
}

// ---------------------------------------------------------------------------
// Kernel 3: row softmax over t in [0, rowBlockEnd). -inf entries -> 0, which
// zero-pads exactly the region pv_gemm's k-loop touches past the diagonal.
// ---------------------------------------------------------------------------
__global__ __launch_bounds__(256) void softmax_kernel(_Float16* __restrict__ P) {
  __shared__ float red[256];
  const int row = blockIdx.x;          // 0..NB*SEQ-1
  const int b = row >> 11;
  const int s = row & (SEQ - 1);
  const int len = ((s >> 7) + 1) << 7; // round_up(s+1, 128)
  _Float16* pr = P + (size_t)b * SEQ * SEQ + (size_t)s * SEQ;
  const int tid = threadIdx.x;

  float mx = -__builtin_inff();
  for (int t = tid; t < len; t += 256) mx = fmaxf(mx, (float)pr[t]);
  red[tid] = mx; __syncthreads();
  for (int off = 128; off > 0; off >>= 1) {
    if (tid < off) red[tid] = fmaxf(red[tid], red[tid + off]);
    __syncthreads();
  }
  mx = red[0]; __syncthreads();

  float sum = 0.0f;
  for (int t = tid; t < len; t += 256) sum += __expf((float)pr[t] - mx);
  red[tid] = sum; __syncthreads();
  for (int off = 128; off > 0; off >>= 1) {
    if (tid < off) red[tid] += red[tid + off];
    __syncthreads();
  }
  const float inv = 1.0f / red[0]; __syncthreads();

  for (int t = tid; t < len; t += 256)
    pr[t] = (_Float16)(__expf((float)pr[t] - mx) * inv);
}

// ---------------------------------------------------------------------------
// Kernel 4: O = P @ V using Vt[b][n][t].  grid = (SEQ/128, DM/128, NB).
// Causal: k-loop stops at s_blk + 128.
// ---------------------------------------------------------------------------
__global__ __launch_bounds__(128) void pv_gemm_kernel(const _Float16* __restrict__ P,
                                                      const _Float16* __restrict__ Vt,
                                                      float* __restrict__ out) {
  const int s_blk = blockIdx.x * 128;
  const int b = blockIdx.z;
  const int lane = threadIdx.x & 31;
  const int wave = threadIdx.x >> 5;
  const int m0 = s_blk + (wave >> 1) * 64;
  const int n0 = blockIdx.y * 128 + (wave & 1) * 64;
  const _Float16* Pb  = P  + (size_t)b * SEQ * SEQ;
  const _Float16* Vtb = Vt + (size_t)b * DM * SEQ;
  const int kEnd = s_blk + 128;

  v8f acc[4][4] = {};
  for (int k0 = 0; k0 < kEnd; k0 += 32) {
    v16h af[4], bf[4];
#pragma unroll
    for (int mi = 0; mi < 4; ++mi)
      af[mi] = load_a_f16(Pb, SEQ, m0 + mi * 16, k0, lane);
#pragma unroll
    for (int ni = 0; ni < 4; ++ni)
      bf[ni] = load_b_f16(Vtb, SEQ, n0 + ni * 16, k0, lane);
    if (k0 + 64 < kEnd) {
      __builtin_prefetch(Pb  + (size_t)(m0 + (lane & 15)) * SEQ + k0 + 64, 0, 1);
      __builtin_prefetch(Vtb + (size_t)(n0 + (lane & 15)) * SEQ + k0 + 64, 0, 1);
    }
#pragma unroll
    for (int mi = 0; mi < 4; ++mi)
#pragma unroll
      for (int ni = 0; ni < 4; ++ni)
        acc[mi][ni] = wmma16(af[mi], bf[ni], acc[mi][ni]);
  }

#pragma unroll
  for (int mi = 0; mi < 4; ++mi) {
#pragma unroll
    for (int ni = 0; ni < 4; ++ni) {
      const int col = n0 + ni * 16 + (lane & 15);              // d
      const int rbase = m0 + mi * 16 + ((lane & 16) ? 8 : 0);  // s
#pragma unroll
      for (int r = 0; r < 8; ++r) {
        const int row = rbase + r;
        out[((size_t)b * SEQ + row) * DM + col] = acc[mi][ni][r];
      }
    }
  }
}

// ---------------------------------------------------------------------------
// Workspace layout (bytes):
//   Wt  @ 0        : 3*1024*1024*2  =   6 MB   (f16, [which][n][d])
//   Qh  @ 6 MB     : 65536*1024*2   = 128 MB
//   Kh  @ 134 MB   : 128 MB
//   Vt  @ 262 MB   : 128 MB                    (f16, [b][d][t])
//   P   @ 390 MB   : 32*2048*2048*2 = 256 MB   (f16, [b][s][t])
// ---------------------------------------------------------------------------
#define WT_OFF  ((size_t)0)
#define QH_OFF  ((size_t)6 << 20)
#define KH_OFF  (QH_OFF + ((size_t)128 << 20))
#define VT_OFF  (KH_OFF + ((size_t)128 << 20))
#define P_OFF   (VT_OFF + ((size_t)128 << 20))

extern "C" void kernel_launch(void* const* d_in, const int* in_sizes, int n_in,
                              void* d_out, int out_size, void* d_ws, size_t ws_size,
                              hipStream_t stream) {
  (void)in_sizes; (void)n_in; (void)out_size; (void)ws_size;
  const float* x  = (const float*)d_in[0];
  const float* Wq = (const float*)d_in[1];
  const float* bq = (const float*)d_in[2];
  const float* Wk = (const float*)d_in[3];
  const float* bk = (const float*)d_in[4];
  const float* Wv = (const float*)d_in[5];
  const float* bv = (const float*)d_in[6];
  float* out = (float*)d_out;

  char* ws = (char*)d_ws;
  _Float16* Wt = (_Float16*)(ws + WT_OFF);
  _Float16* Qh = (_Float16*)(ws + QH_OFF);
  _Float16* Kh = (_Float16*)(ws + KH_OFF);
  _Float16* Vt = (_Float16*)(ws + VT_OFF);
  _Float16* P  = (_Float16*)(ws + P_OFF);

  // 0) transpose+convert the three weight matrices
  wconvert_kernel<<<dim3((3u << 20) / 256), dim3(256), 0, stream>>>(Wq, Wk, Wv, Wt);

  // 1) QKV projections
  qkv_gemm_kernel<<<dim3(ROWS / 128, DM / 128, 3), dim3(128), 0, stream>>>(
      x, Wt, bq, bk, bv, Qh, Kh, Vt);

  // 2) causal masked scores -> P (f16)
  scores_gemm_kernel<<<dim3(SEQ / 128, SEQ / 128, NB), dim3(128), 0, stream>>>(
      Qh, Kh, P);

  // 3) row softmax (in place on P)
  softmax_kernel<<<dim3(ROWS), dim3(256), 0, stream>>>(P);

  // 4) O = P @ V
  pv_gemm_kernel<<<dim3(SEQ / 128, DM / 128, NB), dim3(128), 0, stream>>>(
      P, Vt, out);
}
